// LocalAttention_21827023798427
// MI455X (gfx1250) — matrix operand order repaired and checked by reading, not compiled, not run on hardware
//
#include <hip/hip_runtime.h>
#include <hip/hip_bf16.h>

typedef __attribute__((ext_vector_type(16))) _Float16 v16h;
typedef __attribute__((ext_vector_type(8)))  float    v8f;

#define HD    128
#define HGT   64
#define WID   64
#define TW    16     // pixels per width-tile
#define HALO  18     // TW + 2 halo columns
#define HCOLS 32     // halo columns padded to 32 (WMMA K)
#define QKSCALE 0.08838834764831845f   // 128^-0.5

__launch_bounds__(128)
__global__ void natten3x3_wmma(const float* __restrict__ q,
                               const float* __restrict__ k,
                               const float* __restrict__ v,
                               float* __restrict__ out) {
  // LDS staging (f16 operands, f32 logits). Total ~61 KB of 320 KB/WGP.
  __shared__ _Float16 sQ [TW][HD];        // A operand for QK: [pixel][ch]
  __shared__ _Float16 sKt[3][HCOLS][HD];  // B operand for QK: [krow][haloCol][ch]
  __shared__ _Float16 sVt[3][HD][HCOLS];  // B operand for AV: [krow][ch][haloCol]
  __shared__ float    sS [3][TW][HCOLS];  // logits -> probs (f32)
  __shared__ _Float16 sA [3][TW][HCOLS];  // banded attn (f16), zero off-band

  const int tile = blockIdx.x;            // b * 64 * 4 + h * 4 + wt
  const int w0 = (tile & 3) * TW;
  const int h  = (tile >> 2) & 63;
  const int b  = tile >> 8;

  const int tid  = threadIdx.x;
  const int lane = tid & 31;
  const int wave = tid >> 5;

  const size_t base = (size_t)b * HD * HGT * WID;   // [b][c][h][w] layout

  // ---- Phase 1a: load Q tile (16 px x 128 ch), f32 -> f16
  {
    const int p  = tid & 15;
    const int c0 = tid >> 4;              // 0..7
    for (int it = 0; it < 16; ++it) {
      const int c = c0 + it * 8;
      sQ[p][c] = (_Float16)q[base + ((size_t)c * HGT + h) * WID + w0 + p];
    }
  }
  // ---- Phase 1b: load K halo (col-major in LDS) and V halo (ch-major), zero-pad OOB
  {
    const int col = tid & 31;
    const int c0  = tid >> 5;             // 0..3
    for (int r = 0; r < 3; ++r) {
      const int hh = h + r - 1;
      const bool rowOK = (hh >= 0) && (hh < HGT);
      const int wwi = w0 - 1 + col;
      const bool colOK = (col < HALO) && (wwi >= 0) && (wwi < WID);
      for (int it = 0; it < 32; ++it) {
        const int c = c0 + it * 4;
        float kv = 0.f, vv = 0.f;
        if (rowOK && colOK) {
          const size_t idx = base + ((size_t)c * HGT + hh) * WID + wwi;
          kv = k[idx];
          vv = v[idx];
        }
        sKt[r][col][c] = (_Float16)kv;    // transpose via LDS scatter
        sVt[r][c][col] = (_Float16)vv;    // matches global layout directly
      }
    }
  }
  __syncthreads();

  // Per-lane WMMA operand geometry (ISA 7.12.2, wave32):
  //  A (16x32 f16): M = lane%16; code's contiguous K = t..t+15, t = (lane>=16)*16 (same as B)
  //  B (32x16 f16): N = lane%16; code's contiguous K = t..t+15, t = (lane>=16)*16 (A uses the same)
  const int m   = lane & 15;
  const int s8  = (lane >> 4) * 8;
  const int t16 = (lane >> 4) * 16;

  // ---- Phase 2: QK logits. 6 tiles = 3 kernel-rows x 2 col-tiles, over 4 waves.
  for (int tI = wave; tI < 6; tI += 4) {
    const int r  = tI >> 1;
    const int ct = tI & 1;                // halo cols [ct*16, ct*16+15]
    const int n  = ct * 16 + m;
    v8f acc = {};
    for (int kc = 0; kc < 4; ++kc) {      // 128 channels = 4 x K32
      const int kb = kc * 32;
      v16h a, bm;
      #pragma unroll
      for (int j = 0; j < 8; ++j) {
        a[j]     = sQ[m][kb + t16 + j];
        a[8 + j] = sQ[m][kb + t16 + 8 + j];
      }
      #pragma unroll
      for (int i = 0; i < 16; ++i)
        bm[i] = sKt[r][n][kb + t16 + i];
      acc = __builtin_amdgcn_wmma_f32_16x16x32_f16(false, a, false, bm,
                                                   (short)0, acc, false, false);
    }
    #pragma unroll
    for (int g = 0; g < 8; ++g) {
      const int mm = g + (lane >> 4) * 8; // pixel row of C tile
      sS[r][mm][n] = acc[g] * QKSCALE;
    }
  }
  __syncthreads();

  // ---- Phase 3: softmax over the 9-neighbor band (pixel p -> halo cols p..p+2)
  if (tid < TW) {
    const int p = tid;
    float mx = -1e30f;
    for (int r = 0; r < 3; ++r)
      for (int j = 0; j < 3; ++j)
        mx = fmaxf(mx, sS[r][p][p + j]);
    float e[9], sum = 0.f;
    for (int r = 0; r < 3; ++r)
      for (int j = 0; j < 3; ++j) {
        const float t = __expf(sS[r][p][p + j] - mx);
        e[r * 3 + j] = t;
        sum += t;
      }
    const float inv = 1.f / sum;
    for (int r = 0; r < 3; ++r)
      for (int j = 0; j < 3; ++j)
        sS[r][p][p + j] = e[r * 3 + j] * inv;
  }
  __syncthreads();
  // banded attn matrix in f16 (zeros off-band)
  for (int e = tid; e < 3 * TW * HCOLS; e += 128) {
    const int r   = e >> 9;               // / (16*32)
    const int rem = e & 511;
    const int p   = rem >> 5;
    const int col = rem & 31;
    const float val = (col >= p && col <= p + 2) ? sS[r][p][col] : 0.f;
    sA[r][p][col] = (_Float16)val;
  }
  __syncthreads();

  // ---- Phase 4: AV. 8 channel-group tiles over 4 waves; accumulate 3 kernel rows.
  for (int ng = wave; ng < 8; ng += 4) {
    const int ch = ng * 16 + m;
    v8f acc = {};
    for (int r = 0; r < 3; ++r) {
      v16h a, bm;
      #pragma unroll
      for (int j = 0; j < 8; ++j) {
        a[j]     = sA[r][m][t16 + j];
        a[8 + j] = sA[r][m][t16 + 8 + j];
      }
      #pragma unroll
      for (int i = 0; i < 16; ++i)
        bm[i] = sVt[r][ch][t16 + i];
      acc = __builtin_amdgcn_wmma_f32_16x16x32_f16(false, a, false, bm,
                                                   (short)0, acc, false, false);
    }
    #pragma unroll
    for (int g = 0; g < 8; ++g) {
      const int mm = g + (lane >> 4) * 8; // pixel
      out[(((size_t)b * HGT + h) * WID + (w0 + mm)) * HD + ch] = acc[g];
    }
  }
}

extern "C" void kernel_launch(void* const* d_in, const int* in_sizes, int n_in,
                              void* d_out, int out_size, void* d_ws, size_t ws_size,
                              hipStream_t stream) {
  (void)in_sizes; (void)n_in; (void)d_ws; (void)ws_size; (void)out_size;
  const float* q = (const float*)d_in[0];
  const float* k = (const float*)d_in[1];
  const float* v = (const float*)d_in[2];
  float* out = (float*)d_out;
  dim3 grid(32 * 64 * 4);   // B * H * (W/16)
  dim3 block(128);          // 4 wave32s
  natten3x3_wmma<<<grid, block, 0, stream>>>(q, k, v, out);
}
